// GeneFlowAttention_9182640079431
// MI455X (gfx1250) — matrix operand order repaired and checked by reading, not compiled, not run on hardware
//
#include <hip/hip_runtime.h>
#include <hip/hip_bf16.h>
#include <cstdint>
#include <cstddef>

// Problem constants (match reference)
#define BB 4
#define LL 4096
#define DM 1024
#define HH 16
#define HD 64
#define MM (BB * LL)        // 16384 rows
static constexpr float EPSV = 1e-6f;

typedef __bf16 bf16_t;
typedef bf16_t v16bf __attribute__((ext_vector_type(16)));
typedef float  v8f   __attribute__((ext_vector_type(8)));

union FragBF { v16bf v; uint4 q[2]; };

// fp32 -> bf16 via hardware convert
__device__ __forceinline__ unsigned short f2bf(float f) {
  union { __bf16 b; unsigned short u; } c;
  c.b = (__bf16)f;
  return c.u;
}
__device__ __forceinline__ unsigned int pk2bf(float x, float y) {
  union { __bf16 b[2]; unsigned int u; } c;
  c.b[0] = (__bf16)x;
  c.b[1] = (__bf16)y;
  return c.u;
}
__device__ __forceinline__ float sigmoidf_(float x) {
  return 1.0f / (1.0f + __expf(-x));
}

// Async global->LDS 16-byte copy (tracked by ASYNCcnt).
// LDS flat address low 32 bits == LDS byte offset (ISA aperture rule).
__device__ __forceinline__ void async_cp_b128(const void* gptr, void* lptr) {
  unsigned int lds_off = (unsigned int)(unsigned long long)lptr;
  asm volatile("global_load_async_to_lds_b128 %0, %1, off"
               :: "v"(lds_off), "v"(gptr) : "memory");
}
__device__ __forceinline__ void async_wait0() {
  asm volatile("s_wait_asynccnt 0x0" ::: "memory");
}
__device__ __forceinline__ void async_wait3() {
  asm volatile("s_wait_asynccnt 0x3" ::: "memory");
}

// ---------------------------------------------------------------------------
// Prep: fp32 -> bf16 elementwise, 8 elems/thread
// ---------------------------------------------------------------------------
__global__ __launch_bounds__(256)
void cvt_bf16_kernel(const float* __restrict__ in,
                     unsigned short* __restrict__ outb) {
  size_t i = ((size_t)blockIdx.x * 256 + threadIdx.x) * 8;
  float4 a = *(const float4*)&in[i];
  float4 b = *(const float4*)&in[i + 4];
  uint4 p;
  p.x = pk2bf(a.x, a.y);
  p.y = pk2bf(a.z, a.w);
  p.z = pk2bf(b.x, b.y);
  p.w = pk2bf(b.z, b.w);
  *(uint4*)&outb[i] = p;
}

// ---------------------------------------------------------------------------
// Prep: W[K,N] fp32 -> Wt[N,K] bf16 (transpose so B fragments are K-contig)
// ---------------------------------------------------------------------------
__global__ __launch_bounds__(256)
void prep_w_kernel(const float* __restrict__ W,
                   unsigned short* __restrict__ Wt) {
  int idx = (blockIdx.x * 256 + threadIdx.x) * 4;   // over 1M elements
  int k = idx >> 10;
  int n = idx & 1023;
  float4 f = *(const float4*)&W[(size_t)k * DM + n];
  Wt[(size_t)(n + 0) * DM + k] = f2bf(f.x);
  Wt[(size_t)(n + 1) * DM + k] = f2bf(f.y);
  Wt[(size_t)(n + 2) * DM + k] = f2bf(f.z);
  Wt[(size_t)(n + 3) * DM + k] = f2bf(f.w);
}

// ---------------------------------------------------------------------------
// C[M,N] = act(A_bf16[M,K] @ Wt_bf16[N,K]^T + bias[N]); act 0=sigmoid 1=none
// Block tile 128x64, K-step 32, 8 waves (4x2), each wave 2x2 wmma tiles.
// Double-buffered LDS staging via global_load_async_to_lds_b128:
// tile k+1's copies are in flight while tile k is computed (wait asynccnt<=3).
// ---------------------------------------------------------------------------
__global__ __launch_bounds__(256)
void wmma_gemm_kernel(const unsigned short* __restrict__ A,
                      const unsigned short* __restrict__ Wt,
                      const float* __restrict__ bias, float* __restrict__ C,
                      int Mdim, int Kdim, int Ndim, int act) {
  __shared__ __align__(16) unsigned short As[2][128][32];   // 16 KB
  __shared__ __align__(16) unsigned short Bt[2][64][32];    //  8 KB

  const int tid  = threadIdx.x;
  const int lane = tid & 31;
  const int wid  = tid >> 5;        // 0..7
  const int wm   = wid >> 1;        // 0..3 (M groups of 32)
  const int wn   = wid & 1;         // 0..1 (N groups of 32)
  const int hi   = (lane >> 4) & 1;
  const int ln   = lane & 15;

  const int m0 = blockIdx.x * 128;
  const int n0 = blockIdx.y * 64;

  // Per-thread async-copy coordinates (fixed across k-steps)
  const int a0r = (tid * 16) >> 6;            // A issue 0: row
  const int a0c = ((tid * 16) & 63) >> 1;     //            elem offset
  const int a1r = ((256 + tid) * 16) >> 6;    // A issue 1
  const int a1c = (((256 + tid) * 16) & 63) >> 1;
  const int bnr = (tid * 16) >> 6;            // B issue: row (0..63)
  const int bnc = ((tid * 16) & 63) >> 1;

  auto issue_tile = [&](int buf, int k0) {
    async_cp_b128(&A[(size_t)(m0 + a0r) * Kdim + k0 + a0c], &As[buf][a0r][a0c]);
    async_cp_b128(&A[(size_t)(m0 + a1r) * Kdim + k0 + a1c], &As[buf][a1r][a1c]);
    async_cp_b128(&Wt[(size_t)(n0 + bnr) * Kdim + k0 + bnc], &Bt[buf][bnr][bnc]);
  };

  v8f acc[2][2];
  acc[0][0] = 0.f; acc[0][1] = 0.f; acc[1][0] = 0.f; acc[1][1] = 0.f;

  issue_tile(0, 0);
  int cur = 0;
  for (int k0 = 0; k0 < Kdim; k0 += 32) {
    const bool more = (k0 + 32 < Kdim);
    if (more) issue_tile(cur ^ 1, k0 + 32);     // pipeline next tile
    if (k0 + 64 < Kdim) {
      __builtin_prefetch(&A[(size_t)(m0 + (tid >> 1)) * Kdim + k0 + 64], 0, 1);
    }
    if (more) async_wait3();                    // tile k done, k+1 in flight
    else      async_wait0();
    __syncthreads();

    FragBF afrag[2], bfrag[2];
#pragma unroll
    for (int ms = 0; ms < 2; ++ms) {
      int row = wm * 32 + ms * 16 + ln;
      afrag[ms].q[0] = *(const uint4*)&As[cur][row][hi * 8];
      afrag[ms].q[1] = *(const uint4*)&As[cur][row][16 + hi * 8];
    }
#pragma unroll
    for (int ns = 0; ns < 2; ++ns) {
      int col = wn * 32 + ns * 16 + ln;
      bfrag[ns].q[0] = *(const uint4*)&Bt[cur][col][hi * 8];
      bfrag[ns].q[1] = *(const uint4*)&Bt[cur][col][16 + hi * 8];
    }
#pragma unroll
    for (int ms = 0; ms < 2; ++ms)
#pragma unroll
      for (int ns = 0; ns < 2; ++ns)
        acc[ms][ns] = __builtin_amdgcn_wmma_f32_16x16x32_bf16(
            false, afrag[ms].v, false, bfrag[ns].v, (short)0, acc[ms][ns],
            false, false);
    __syncthreads();
    cur ^= 1;
  }

#pragma unroll
  for (int ms = 0; ms < 2; ++ms) {
#pragma unroll
    for (int ns = 0; ns < 2; ++ns) {
      int col = n0 + wn * 32 + ns * 16 + ln;
      float bvv = bias ? bias[col] : 0.0f;
#pragma unroll
      for (int r = 0; r < 8; ++r) {
        int row = m0 + wm * 32 + ms * 16 + r + hi * 8;
        float v = acc[ms][ns][r] + bvv;
        if (act == 0) v = sigmoidf_(v);
        C[(size_t)row * Ndim + col] = v;
      }
    }
  }
}

// ---------------------------------------------------------------------------
// Column sums over L (optionally weighted)
// ---------------------------------------------------------------------------
__global__ __launch_bounds__(256)
void colsum_kernel(const float* __restrict__ qf, const float* __restrict__ kf,
                   const float* __restrict__ si, const float* __restrict__ so,
                   float* __restrict__ qout, float* __restrict__ kout,
                   int weighted) {
  __shared__ float sq[256], sk[256];
  int bh = blockIdx.x; int b = bh / HH; int h = bh % HH;
  int t = threadIdx.x; int d = t & 63; int part = t >> 6;
  float aq = 0.f, ak = 0.f;
  for (int l = part * (LL / 4); l < (part + 1) * (LL / 4); ++l) {
    size_t idx = ((size_t)(b * LL + l)) * DM + h * HD + d;
    float wq = 1.f, wk = 1.f;
    if (weighted) { wq = si[bh * LL + l]; wk = so[bh * LL + l]; }
    aq += qf[idx] * wq;
    ak += kf[idx] * wk;
  }
  sq[t] = aq; sk[t] = ak;
  __syncthreads();
  if (part == 0) {
    qout[bh * HD + d] = sq[d] + sq[64 + d] + sq[128 + d] + sq[192 + d];
    kout[bh * HD + d] = sk[d] + sk[64 + d] + sk[128 + d] + sk[192 + d];
  }
}

// ---------------------------------------------------------------------------
// Per-token pass 1: sink_incoming / source_outgoing reciprocal dots
// ---------------------------------------------------------------------------
__global__ __launch_bounds__(256)
void perl1_kernel(const float* __restrict__ qf, const float* __restrict__ kf,
                  const float* __restrict__ ksum, const float* __restrict__ qsum,
                  float* __restrict__ si, float* __restrict__ so) {
  int gid = blockIdx.x * 256 + threadIdx.x;   // = bh*LL + l
  int bh = gid / LL; int l = gid % LL;
  int b = bh / HH; int h = bh % HH;
  size_t base = ((size_t)(b * LL + l)) * DM + h * HD;
  float accs = 0.f, acco = 0.f;
#pragma unroll 4
  for (int d = 0; d < HD; ++d) {
    accs += (qf[base + d] + EPSV) * (ksum[bh * HD + d] + EPSV);
    acco += (kf[base + d] + EPSV) * (qsum[bh * HD + d] + EPSV);
  }
  si[gid] = 1.0f / accs;
  so[gid] = 1.0f / acco;
}

// ---------------------------------------------------------------------------
// Per-token pass 2: conserved sink (-> sigmoid alloc) and clipped source
// ---------------------------------------------------------------------------
__global__ __launch_bounds__(256)
void perl2_kernel(const float* __restrict__ qf, const float* __restrict__ kf,
                  const float* __restrict__ kw, const float* __restrict__ qw,
                  float* __restrict__ salloc, float* __restrict__ csrc) {
  int gid = blockIdx.x * 256 + threadIdx.x;
  int bh = gid / LL; int l = gid % LL;
  int b = bh / HH; int h = bh % HH;
  size_t base = ((size_t)(b * LL + l)) * DM + h * HD;
  float cs = 0.f, cc = 0.f;
#pragma unroll 4
  for (int d = 0; d < HD; ++d) {
    cs += (qf[base + d] + EPSV) * (kw[bh * HD + d] + EPSV);
    cc += (kf[base + d] + EPSV) * (qw[bh * HD + d] + EPSV);
  }
  salloc[gid] = sigmoidf_(cs);
  csrc[gid] = fminf(1.0f, fmaxf(-1.0f, cc));
}

// ---------------------------------------------------------------------------
// Softmax over L per (b,h), scaled by L
// ---------------------------------------------------------------------------
__global__ __launch_bounds__(256)
void softmax_kernel(const float* __restrict__ csrc, float* __restrict__ comp) {
  __shared__ float red[256];
  int bh = blockIdx.x; int t = threadIdx.x;
  const float* xrow = csrc + (size_t)bh * LL;
  float mx = -1e30f;
  for (int l = t; l < LL; l += 256) mx = fmaxf(mx, xrow[l]);
  red[t] = mx; __syncthreads();
  for (int s = 128; s > 0; s >>= 1) {
    if (t < s) red[t] = fmaxf(red[t], red[t + s]);
    __syncthreads();
  }
  mx = red[0]; __syncthreads();
  float sum = 0.f;
  for (int l = t; l < LL; l += 256) sum += __expf(xrow[l] - mx);
  red[t] = sum; __syncthreads();
  for (int s = 128; s > 0; s >>= 1) {
    if (t < s) red[t] += red[t + s];
    __syncthreads();
  }
  sum = red[0];
  float scale = (float)LL / sum;
  for (int l = t; l < LL; l += 256)
    comp[(size_t)bh * LL + l] = __expf(xrow[l] - mx) * scale;
}

// ---------------------------------------------------------------------------
// kv[bh,d,m] = sum_l k[bh,l,d] * v[bh,l,m] * comp[bh,l]  (bf16 WMMA, K=L)
// ---------------------------------------------------------------------------
__global__ __launch_bounds__(256)
void kvstate_kernel(const float* __restrict__ kf, const float* __restrict__ vf,
                    const float* __restrict__ comp, float* __restrict__ kv) {
  __shared__ __align__(16) unsigned short kT[64][32];
  __shared__ __align__(16) unsigned short vT[64][32];
  int bh = blockIdx.x; int b = bh / HH; int h = bh % HH;
  int tid = threadIdx.x; int lane = tid & 31; int wid = tid >> 5;
  int tm = wid >> 1;
  int wn = wid & 1;
  int hi = (lane >> 4) & 1; int ln = lane & 15;

  v8f acc[2]; acc[0] = 0.f; acc[1] = 0.f;

  for (int l0 = 0; l0 < LL; l0 += 32) {
#pragma unroll
    for (int i = 0; i < 2; ++i) {
      int idx = (i * 256 + tid) * 4;
      int li = idx >> 6; int d = idx & 63;
      size_t g = ((size_t)(b * LL + l0 + li)) * DM + h * HD + d;
      float4 fk = *(const float4*)&kf[g];
      float4 fv = *(const float4*)&vf[g];
      float c = comp[bh * LL + l0 + li];
      kT[d + 0][li] = f2bf(fk.x); kT[d + 1][li] = f2bf(fk.y);
      kT[d + 2][li] = f2bf(fk.z); kT[d + 3][li] = f2bf(fk.w);
      vT[d + 0][li] = f2bf(fv.x * c); vT[d + 1][li] = f2bf(fv.y * c);
      vT[d + 2][li] = f2bf(fv.z * c); vT[d + 3][li] = f2bf(fv.w * c);
    }
    __syncthreads();
    FragBF af, bf0, bf1;
    int row = tm * 16 + ln;
    af.q[0] = *(const uint4*)&kT[row][hi * 8];
    af.q[1] = *(const uint4*)&kT[row][16 + hi * 8];
    int c0 = wn * 32 + ln;
    bf0.q[0] = *(const uint4*)&vT[c0][hi * 8];
    bf0.q[1] = *(const uint4*)&vT[c0][16 + hi * 8];
    int c1 = wn * 32 + 16 + ln;
    bf1.q[0] = *(const uint4*)&vT[c1][hi * 8];
    bf1.q[1] = *(const uint4*)&vT[c1][16 + hi * 8];
    acc[0] = __builtin_amdgcn_wmma_f32_16x16x32_bf16(false, af.v, false, bf0.v,
                                                     (short)0, acc[0], false, false);
    acc[1] = __builtin_amdgcn_wmma_f32_16x16x32_bf16(false, af.v, false, bf1.v,
                                                     (short)0, acc[1], false, false);
    __syncthreads();
  }
#pragma unroll
  for (int ns = 0; ns < 2; ++ns) {
    int col = wn * 32 + ns * 16 + ln;
#pragma unroll
    for (int r = 0; r < 8; ++r) {
      int drow = tm * 16 + r + hi * 8;
      kv[((size_t)bh * HD + drow) * HD + col] = acc[ns][r];
    }
  }
}

// ---------------------------------------------------------------------------
// attn_bf[b,l,h*64+m] = bf16( salloc * sum_d (q*si)[l,d] * kv[d,m] )
// ---------------------------------------------------------------------------
__global__ __launch_bounds__(256)
void applykv_kernel(const float* __restrict__ qf, const float* __restrict__ si,
                    const float* __restrict__ salloc,
                    const float* __restrict__ kv,
                    unsigned short* __restrict__ attn_bf) {
  __shared__ __align__(16) unsigned short As[128][32];
  __shared__ __align__(16) unsigned short Bt[64][32];
  int bh = blockIdx.y; int b = bh / HH; int h = bh % HH;
  int l0 = blockIdx.x * 128;
  int tid = threadIdx.x; int lane = tid & 31; int wid = tid >> 5;
  int wm = wid >> 1; int wn = wid & 1;
  int hi = (lane >> 4) & 1; int ln = lane & 15;

  v8f acc[2][2];
  acc[0][0] = 0.f; acc[0][1] = 0.f; acc[1][0] = 0.f; acc[1][1] = 0.f;

  for (int k0 = 0; k0 < HD; k0 += 32) {
#pragma unroll
    for (int i = 0; i < 4; ++i) {
      int idx = (i * 256 + tid) * 4;
      int r = idx >> 5; int c = idx & 31;
      size_t g = ((size_t)(b * LL + l0 + r)) * DM + h * HD + k0 + c;
      float4 f = *(const float4*)&qf[g];
      float w = si[bh * LL + l0 + r];
      uint2 p;
      p.x = pk2bf(f.x * w, f.y * w);
      p.y = pk2bf(f.z * w, f.w * w);
      *(uint2*)&As[r][c] = p;
    }
#pragma unroll
    for (int i = 0; i < 2; ++i) {
      int idx = (i * 256 + tid) * 4;
      int kk = idx >> 6; int n = idx & 63;
      float4 f = *(const float4*)&kv[((size_t)bh * HD + k0 + kk) * HD + n];
      Bt[n + 0][kk] = f2bf(f.x); Bt[n + 1][kk] = f2bf(f.y);
      Bt[n + 2][kk] = f2bf(f.z); Bt[n + 3][kk] = f2bf(f.w);
    }
    __syncthreads();

    FragBF afrag[2], bfrag[2];
#pragma unroll
    for (int ms = 0; ms < 2; ++ms) {
      int row = wm * 32 + ms * 16 + ln;
      afrag[ms].q[0] = *(const uint4*)&As[row][hi * 8];
      afrag[ms].q[1] = *(const uint4*)&As[row][16 + hi * 8];
    }
#pragma unroll
    for (int ns = 0; ns < 2; ++ns) {
      int col = wn * 32 + ns * 16 + ln;
      bfrag[ns].q[0] = *(const uint4*)&Bt[col][hi * 8];
      bfrag[ns].q[1] = *(const uint4*)&Bt[col][16 + hi * 8];
    }
#pragma unroll
    for (int ms = 0; ms < 2; ++ms)
#pragma unroll
      for (int ns = 0; ns < 2; ++ns)
        acc[ms][ns] = __builtin_amdgcn_wmma_f32_16x16x32_bf16(
            false, afrag[ms].v, false, bfrag[ns].v, (short)0, acc[ms][ns],
            false, false);
    __syncthreads();
  }

#pragma unroll
  for (int ms = 0; ms < 2; ++ms) {
#pragma unroll
    for (int ns = 0; ns < 2; ++ns) {
      int col = wn * 32 + ns * 16 + ln;
#pragma unroll
      for (int r = 0; r < 8; ++r) {
        int lrow = l0 + wm * 32 + ms * 16 + r + hi * 8;
        float sa = salloc[bh * LL + lrow];
        attn_bf[((size_t)(b * LL + lrow)) * DM + h * HD + col] =
            f2bf(acc[ms][ns][r] * sa);
      }
    }
  }
}

// ---------------------------------------------------------------------------
extern "C" void kernel_launch(void* const* d_in, const int* in_sizes, int n_in,
                              void* d_out, int out_size, void* d_ws, size_t ws_size,
                              hipStream_t stream) {
  const float* x  = (const float*)d_in[0];
  const float* Wq = (const float*)d_in[1];
  const float* bq = (const float*)d_in[2];
  const float* Wk = (const float*)d_in[3];
  const float* bk = (const float*)d_in[4];
  const float* Wv = (const float*)d_in[5];
  const float* bv = (const float*)d_in[6];
  const float* Wo = (const float*)d_in[7];
  const float* bo = (const float*)d_in[8];
  float* out = (float*)d_out;

  const size_t big = (size_t)MM * DM;   // 16,777,216 elements
  char* base = (char*)d_ws;
  float* qf   = (float*)base;                  base += big * 4;
  float* kf   = (float*)base;                  base += big * 4;
  float* vf   = (float*)base;                  base += big * 4;
  unsigned short* xbf   = (unsigned short*)base; base += big * 2;
  unsigned short* attnb = (unsigned short*)base; base += big * 2;
  unsigned short* wtq = (unsigned short*)base; base += (size_t)DM * DM * 2;
  unsigned short* wtk = (unsigned short*)base; base += (size_t)DM * DM * 2;
  unsigned short* wtv = (unsigned short*)base; base += (size_t)DM * DM * 2;
  unsigned short* wto = (unsigned short*)base; base += (size_t)DM * DM * 2;
  float* qsum = (float*)base; base += BB * HH * HD * 4;
  float* ksum = (float*)base; base += BB * HH * HD * 4;
  float* qw   = (float*)base; base += BB * HH * HD * 4;
  float* kw   = (float*)base; base += BB * HH * HD * 4;
  float* si   = (float*)base; base += (size_t)BB * HH * LL * 4;
  float* so   = (float*)base; base += (size_t)BB * HH * LL * 4;
  float* csrc = (float*)base; base += (size_t)BB * HH * LL * 4;
  float* sal  = (float*)base; base += (size_t)BB * HH * LL * 4;
  float* comp = (float*)base; base += (size_t)BB * HH * LL * 4;
  float* kv   = (float*)base; base += (size_t)BB * HH * HD * HD * 4;

  dim3 blk(256);
  dim3 gg(MM / 128, DM / 64);

  // One-time precision conversion / weight transpose
  cvt_bf16_kernel<<<(unsigned)(big / 2048), blk, 0, stream>>>(x, xbf);
  prep_w_kernel<<<(DM * DM) / 1024, blk, 0, stream>>>(Wq, wtq);
  prep_w_kernel<<<(DM * DM) / 1024, blk, 0, stream>>>(Wk, wtk);
  prep_w_kernel<<<(DM * DM) / 1024, blk, 0, stream>>>(Wv, wtv);
  prep_w_kernel<<<(DM * DM) / 1024, blk, 0, stream>>>(Wo, wto);

  // Projections (Q,K fused sigmoid; V identity)
  wmma_gemm_kernel<<<gg, blk, 0, stream>>>(xbf, wtq, bq, qf, MM, DM, DM, 0);
  wmma_gemm_kernel<<<gg, blk, 0, stream>>>(xbf, wtk, bk, kf, MM, DM, DM, 0);
  wmma_gemm_kernel<<<gg, blk, 0, stream>>>(xbf, wtv, bv, vf, MM, DM, DM, 1);

  // Flow-conservation scalars
  colsum_kernel<<<BB * HH, blk, 0, stream>>>(qf, kf, nullptr, nullptr,
                                             qsum, ksum, 0);
  perl1_kernel<<<(BB * HH * LL) / 256, blk, 0, stream>>>(qf, kf, ksum, qsum,
                                                         si, so);
  colsum_kernel<<<BB * HH, blk, 0, stream>>>(qf, kf, si, so, qw, kw, 1);
  perl2_kernel<<<(BB * HH * LL) / 256, blk, 0, stream>>>(qf, kf, kw, qw,
                                                         sal, csrc);
  softmax_kernel<<<BB * HH, blk, 0, stream>>>(csrc, comp);

  // Linear attention
  kvstate_kernel<<<BB * HH, blk, 0, stream>>>(kf, vf, comp, kv);
  applykv_kernel<<<dim3(LL / 128, BB * HH), blk, 0, stream>>>(qf, si, sal,
                                                              kv, attnb);

  // Output projection (bf16 attn input)
  wmma_gemm_kernel<<<gg, blk, 0, stream>>>(attnb, wto, bo, out, MM, DM, DM, 1);
}
